// MultiHeadAttentionROM_35897336660453
// MI455X (gfx1250) — matrix operand order repaired and checked by reading, not compiled, not run on hardware
//
#include <hip/hip_runtime.h>
#include <hip/hip_bf16.h>

// MI455X / gfx1250, wave32. Linear attention restructured as:
//   KV_bh   = K_bh^T V_bh              (64x64 per head, split-S atomic reduce)
//   Mt_b    = (blockdiag(KV) @ W^T)^T  (bf16, transposed: Mt[o][din], 8MB)
//   out_b   = Q_b @ Mt_b^T + bias      (barrier-free bf16-WMMA GEMM over Q)
// HBM traffic ~= Q+K+V+out+W+Mt ~= 276MB (roofline ~12us); ~43 GF bf16 WMMA.

typedef __attribute__((ext_vector_type(16))) __bf16 v16bf;
typedef __attribute__((ext_vector_type(8)))  __bf16 v8bf;
typedef __attribute__((ext_vector_type(8)))  float  v8f;

#define BSZ    4
#define SSZ    4096
#define DMODEL 1024
#define HNUM   16
#define HDIM   64
#define SPLIT  8   // S-direction split for the KV reduction

static __device__ __forceinline__ v16bf cat16(v8bf lo, v8bf hi) {
  v16bf r;
#pragma unroll
  for (int i = 0; i < 8; ++i) { r[i] = lo[i]; r[i + 8] = hi[i]; }
  return r;
}

static __device__ __forceinline__ v16bf cvt16(float4 f0, float4 f1, float4 f2, float4 f3) {
  v16bf r;
  r[0]  = (__bf16)f0.x; r[1]  = (__bf16)f0.y; r[2]  = (__bf16)f0.z; r[3]  = (__bf16)f0.w;
  r[4]  = (__bf16)f1.x; r[5]  = (__bf16)f1.y; r[6]  = (__bf16)f1.z; r[7]  = (__bf16)f1.w;
  r[8]  = (__bf16)f2.x; r[9]  = (__bf16)f2.y; r[10] = (__bf16)f2.z; r[11] = (__bf16)f2.w;
  r[12] = (__bf16)f3.x; r[13] = (__bf16)f3.y; r[14] = (__bf16)f3.z; r[15] = (__bf16)f3.w;
  return r;
}

// ---------------------------------------------------------------------------
// Kernel 0: zero the KV accumulator region of the workspace (poisoned 0xAA).
// ---------------------------------------------------------------------------
__global__ void __launch_bounds__(256) kv_zero_kernel(float* __restrict__ kv) {
  kv[(size_t)blockIdx.x * blockDim.x + threadIdx.x] = 0.0f;
}

// ---------------------------------------------------------------------------
// Kernel 1: KV[b,h] = K_h^T @ V_h  (64x64), split-S with atomic f32 reduce.
// 512 blocks x 512 threads (16 waves). Each wave owns one 16x16 tile.
// Pure one-pass read of K and V (128MB) -> bandwidth-bound; 512 blocks keep
// all WGPs streaming.
// ---------------------------------------------------------------------------
__global__ void __launch_bounds__(512)
kv_kernel(const float* __restrict__ K, const float* __restrict__ V,
          float* __restrict__ kv) {
  __shared__ __align__(16) __bf16 Kt[HDIM][32];  // transposed: Kt[d][s]
  __shared__ __align__(16) __bf16 Vt[HDIM][32];  // transposed: Vt[e][s]

  const int bh    = blockIdx.x >> 3;   // 0..63
  const int split = blockIdx.x & 7;
  const int b = bh >> 4, h = bh & 15;
  const int tid  = threadIdx.x;
  const int wave = tid >> 5, lane = tid & 31;
  const int ti = wave >> 2, tj = wave & 3;    // tile coords in 4x4 grid
  const int lm = lane & 15, hl = lane >> 4;

  const size_t base = ((size_t)b * SSZ) * DMODEL + (size_t)h * HDIM;
  v8f acc = {};

  const int s_begin = split * (SSZ / SPLIT);
  const int s_end   = s_begin + (SSZ / SPLIT);
  // cooperative-load mapping: each thread moves 4 consecutive f32 per matrix
  const int sl = (tid * 4) >> 6;   // 0..31  (s within chunk)
  const int dl = (tid * 4) & 63;   // 0..60  (feature, multiple of 4)

  for (int s0 = s_begin; s0 < s_end; s0 += 32) {
    const float4 kf = *(const float4*)(K + base + (size_t)(s0 + sl) * DMODEL + dl);
    const float4 vf = *(const float4*)(V + base + (size_t)(s0 + sl) * DMODEL + dl);
    Kt[dl + 0][sl] = (__bf16)kf.x; Kt[dl + 1][sl] = (__bf16)kf.y;
    Kt[dl + 2][sl] = (__bf16)kf.z; Kt[dl + 3][sl] = (__bf16)kf.w;
    Vt[dl + 0][sl] = (__bf16)vf.x; Vt[dl + 1][sl] = (__bf16)vf.y;
    Vt[dl + 2][sl] = (__bf16)vf.z; Vt[dl + 3][sl] = (__bf16)vf.w;
    __syncthreads();

    // A fragment (16x32 bf16): lanes 0-15 hold K=0-7 & 16-23, lanes 16-31 K=8-15 & 24-31
    const int koff = hl * 8;
    v8bf alo = *(const v8bf*)&Kt[ti * 16 + lm][koff];
    v8bf ahi = *(const v8bf*)&Kt[ti * 16 + lm][16 + koff];
    // B fragment (32x16 bf16): lanes 0-15 hold K=0-7 & 16-23, lanes 16-31 K=8-15 & 24-31
    const int k0 = hl * 8;
    v8bf blo = *(const v8bf*)&Vt[tj * 16 + lm][k0];
    v8bf bhi = *(const v8bf*)&Vt[tj * 16 + lm][k0 + 16];

    acc = __builtin_amdgcn_wmma_f32_16x16x32_bf16(
        false, cat16(alo, ahi), false, cat16(blo, bhi), (short)0, acc, false, false);
    __syncthreads();
  }

  float* dst = kv + (size_t)bh * HDIM * HDIM;
#pragma unroll
  for (int r = 0; r < 8; ++r) {
    const int m = r + 8 * hl;
    atomicAdd(&dst[(size_t)(ti * 16 + m) * HDIM + tj * 16 + lm], acc[r]);
  }
}

// ---------------------------------------------------------------------------
// Kernel 2: Mt[b][o][h*64+d] = sum_e KV[b,h,d,e] * W[o][h*64+e]   (bf16,
// TRANSPOSED output). 512 blocks x 256 threads (8 waves), no LDS: both
// fragment k-runs are contiguous (KV rows / W rows), and in the transposed
// layout one lane's 8 accumulator rows are 8 consecutive din values -> one
// packed 16B bf16 store per tile.
// ---------------------------------------------------------------------------
__global__ void __launch_bounds__(256)
proj_kernel(const float* __restrict__ kv, const float* __restrict__ W,
            __bf16* __restrict__ Mt) {
  const int bh = blockIdx.x >> 3;   // 0..63
  const int cb = blockIdx.x & 7;    // 128-col block of the 1024 outputs
  const int b = bh >> 4, h = bh & 15;
  const int wave = threadIdx.x >> 5, lane = threadIdx.x & 31;
  const int lm = lane & 15, hl = lane >> 4;
  const int ocol = cb * 128 + wave * 16 + lm;

  const float* kvb = kv + (size_t)bh * HDIM * HDIM;
  v8f acc[4] = {};

#pragma unroll
  for (int kc = 0; kc < HDIM; kc += 32) {
    // B fragment: B[k][n] = W[ocol][h*64 + kc + k]; k-run contiguous in W row
    const float* wrow = W + (size_t)ocol * DMODEL + h * HDIM + kc + hl * 8;
    const v16bf bfrag = cvt16(*(const float4*)(wrow),
                              *(const float4*)(wrow + 4),
                              *(const float4*)(wrow + 16),
                              *(const float4*)(wrow + 20));
#pragma unroll
    for (int i = 0; i < 4; ++i) {
      const float* arow = kvb + (size_t)(i * 16 + lm) * HDIM + kc + hl * 8;
      const v16bf afrag = cvt16(*(const float4*)(arow),
                                *(const float4*)(arow + 4),
                                *(const float4*)(arow + 16),
                                *(const float4*)(arow + 20));
      acc[i] = __builtin_amdgcn_wmma_f32_16x16x32_bf16(
          false, afrag, false, bfrag, (short)0, acc[i], false, false);
    }
  }

  // transposed bf16 store: Mt[b][ocol][h*64 + i*16 + (r + 8*hl)]
  __bf16* Mb = Mt + ((size_t)b * DMODEL + ocol) * DMODEL;
#pragma unroll
  for (int i = 0; i < 4; ++i) {
    v8bf pk;
#pragma unroll
    for (int r = 0; r < 8; ++r) pk[r] = (__bf16)acc[i][r];
    *(v8bf*)&Mb[h * HDIM + i * 16 + 8 * hl] = pk;
  }
}

// ---------------------------------------------------------------------------
// Kernel 3: out[b] = Q[b] (4096x1024) @ Mt[b]^T (1024x1024 bf16) + bias
// 1024 blocks x 256 threads, NO LDS and NO barriers: both operands stream
// straight from global (Q via cvt, Mt already bf16 with contiguous k-runs),
// so the loop software-pipelines on LOADcnt alone. 128x128 tile per block,
// 8 waves x (64x32) accumulators, 256 v_wmma per wave.
// ---------------------------------------------------------------------------
__global__ void __launch_bounds__(256)
out_kernel(const float* __restrict__ Q, const __bf16* __restrict__ Mt,
           const float* __restrict__ bias, float* __restrict__ out) {
  const int b    = blockIdx.x >> 8;
  const int rest = blockIdx.x & 255;
  const int row0 = (rest >> 3) * 128;
  const int col0 = (rest & 7) * 128;
  const int wave = threadIdx.x >> 5, lane = threadIdx.x & 31;
  const int wr = wave >> 2, wc = wave & 3;      // 2x4 wave grid
  const int lm = lane & 15, hl = lane >> 4;

  // Per-lane base pointers; every fragment k-run is contiguous from these.
  const float* qa[4];
#pragma unroll
  for (int p = 0; p < 4; ++p)
    qa[p] = Q + ((size_t)b * SSZ + row0 + wr * 64 + p * 16 + lm) * DMODEL + hl * 8;
  const __bf16* mb[2];
#pragma unroll
  for (int q = 0; q < 2; ++q)
    mb[q] = Mt + ((size_t)b * DMODEL + col0 + wc * 32 + q * 16 + lm) * DMODEL + hl * 8;

  v8f acc[4][2] = {};

#pragma unroll 2
  for (int kk = 0; kk < DMODEL; kk += 32) {
    // B fragments: two 8-bf16 runs per lane (k 8h..8h+7, 16+8h..) (already transposed+packed)
    v16bf bf[2];
#pragma unroll
    for (int q = 0; q < 2; ++q)
      bf[q] = cat16(*(const v8bf*)(mb[q] + kk), *(const v8bf*)(mb[q] + kk + 16));
    // A fragments: contiguous f32 runs in a Q row, convert to bf16 in regs
#pragma unroll
    for (int p = 0; p < 4; ++p) {
      const float* qp = qa[p] + kk;
      const v16bf af = cvt16(*(const float4*)(qp),
                             *(const float4*)(qp + 4),
                             *(const float4*)(qp + 16),
                             *(const float4*)(qp + 20));
#pragma unroll
      for (int q = 0; q < 2; ++q)
        acc[p][q] = __builtin_amdgcn_wmma_f32_16x16x32_bf16(
            false, af, false, bf[q], (short)0, acc[p][q], false, false);
    }
    if (kk + 32 < DMODEL) {           // global_prefetch_b8 of the next chunk
      __builtin_prefetch(qa[0] + kk + 32, 0, 0);
      __builtin_prefetch(mb[0] + kk + 32, 0, 0);
    }
  }

  // epilogue: fused bias + store (lanes of a tile cover 16 consecutive cols)
#pragma unroll
  for (int q = 0; q < 2; ++q) {
    const int col = col0 + wc * 32 + q * 16 + lm;
    const float bv = bias[col];
#pragma unroll
    for (int p = 0; p < 4; ++p) {
      const int rowb = row0 + wr * 64 + p * 16 + 8 * hl;
#pragma unroll
      for (int r = 0; r < 8; ++r)
        out[((size_t)b * SSZ + rowb + r) * DMODEL + col] = acc[p][q][r] + bv;
    }
  }
}

// ---------------------------------------------------------------------------
extern "C" void kernel_launch(void* const* d_in, const int* in_sizes, int n_in,
                              void* d_out, int out_size, void* d_ws, size_t ws_size,
                              hipStream_t stream) {
  const float* Q    = (const float*)d_in[0];
  const float* K    = (const float*)d_in[1];
  const float* V    = (const float*)d_in[2];
  const float* W    = (const float*)d_in[3];
  const float* bias = (const float*)d_in[4];
  float* out = (float*)d_out;

  float*  kv = (float*)d_ws;                                    // 1 MB f32
  __bf16* Mt = (__bf16*)(kv + (size_t)BSZ * HNUM * HDIM * HDIM); // 8 MB bf16

  kv_zero_kernel<<<(BSZ * HNUM * HDIM * HDIM) / 256, 256, 0, stream>>>(kv);
  kv_kernel<<<BSZ * HNUM * SPLIT, 512, 0, stream>>>(K, V, kv);
  proj_kernel<<<BSZ * HNUM * 8, 256, 0, stream>>>(kv, W, Mt);
  out_kernel<<<BSZ * 256, 256, 0, stream>>>(Q, Mt, bias, out);
}
